// CausalAttention_40192303956712
// MI455X (gfx1250) — matrix operand order repaired and checked by reading, not compiled, not run on hardware
//
#include <hip/hip_runtime.h>
#include <hip/hip_bf16.h>

typedef __attribute__((ext_vector_type(16))) _Float16 v16h;
typedef __attribute__((ext_vector_type(8)))  _Float16 v8h;
typedef __attribute__((ext_vector_type(8)))  float    v8f;
typedef __attribute__((ext_vector_type(4)))  float    v4f;

#define LDA 40   // padded LDS row stride in halves (32 data + 8 pad), 80B = 5*16B

#if defined(__gfx1250__) && __has_builtin(__builtin_amdgcn_tensor_load_to_lds) && \
    __has_builtin(__builtin_amdgcn_s_wait_tensorcnt)
#define USE_TDM 1
typedef __attribute__((ext_vector_type(4))) unsigned int v4u;
typedef __attribute__((ext_vector_type(8))) int          v8i;
typedef __attribute__((ext_vector_type(4))) int          v4i;
#endif

__device__ __forceinline__ v8f wmma_f16(v16h a, v16h b, v8f c) {
    return __builtin_amdgcn_wmma_f32_16x16x32_f16(false, a, false, b, (short)0, c, false, false);
}

__device__ __forceinline__ v16h cat8(v8h lo, v8h hi) {
    v16h r;
#pragma unroll
    for (int i = 0; i < 8; ++i) { r[i] = lo[i]; r[i + 8] = hi[i]; }
    return r;
}

// 16 contiguous halves -> v16h (element j corresponds to offset j)
__device__ __forceinline__ v16h ld16h(const _Float16* p) {
    return cat8(*(const v8h*)p, *(const v8h*)(p + 8));
}

// A-matrix fragment (16x32 f16): lane row m = lane&15;
// element j holds A[m][ (lane&16) + j ]
__device__ __forceinline__ v16h load_a_frag(const _Float16* base, int lda, int lane) {
    int m   = lane & 15;
    int klo = lane & 16;                   // 0 or 16
    const _Float16* p = base + m * lda + klo;
    return ld16h(p);
}

// B-matrix fragment (32x16 f16) from LDS tile stored transposed [n][k]:
// lane col n = lane&15; element j holds B[ ((lane&16)?16:0) + j ][ n ]
__device__ __forceinline__ v16h load_b_frag(const _Float16* base, int ldb, int lane) {
    int n  = lane & 15;
    int kb = lane & 16;                    // 0 or 16
    return ld16h(base + n * ldb + kb);
}

// ---------------------------------------------------------------------------
// Kernel 1: fused QKV projection.
// X[4096,1024]f32 x W[1024, 3*1024]f32 -> Q,K (f16 [b,n,s,h]) and V (f16 [b,n,h,s])
// ---------------------------------------------------------------------------
__global__ __launch_bounds__(256) void qkv_proj_kernel(
    const float* __restrict__ X,
    const float* __restrict__ WQ, const float* __restrict__ WK, const float* __restrict__ WV,
    const float* __restrict__ bQ, const float* __restrict__ bK, const float* __restrict__ bV,
    _Float16* __restrict__ Qws, _Float16* __restrict__ Kws, _Float16* __restrict__ Vt)
{
    __shared__ __align__(16) _Float16 As[128 * LDA];
    __shared__ __align__(16) _Float16 Bt[64 * LDA];

    const int tid = threadIdx.x, lane = tid & 31, wave = tid >> 5;
    const int ct = blockIdx.x;                  // 0..47 : 64-wide column tile
    const int w  = ct >> 4;                     // 0=Q 1=K 2=V
    const int nh = ct & 15;                     // head
    const float* W    = (w == 0) ? WQ : (w == 1) ? WK : WV;
    const float* bias = (w == 0) ? bQ : (w == 1) ? bK : bV;
    const int rowbase = blockIdx.y * 128;
    const int wm = wave & 3, wn = wave >> 2;    // 4x2 wave grid, 32x32 per wave

    v8f acc[2][2] = {};

    const int ar = tid >> 1, ah = (tid & 1) * 16;     // A staging: 16 f32 per thread
    const int bh = tid & 63, bd = (tid >> 6) * 8;     // B staging: 8 f32 per thread

    for (int kk = 0; kk < 32; ++kk) {
        const int kbase = kk * 32;
        __syncthreads();
        {   // stage A tile 128x32 (f32 -> f16)
            const float* xp = X + (size_t)(rowbase + ar) * 1024 + kbase + ah;
            if (kk + 1 < 32) __builtin_prefetch(xp + 32, 0, 0);
            v4f x0 = *(const v4f*)xp,       x1 = *(const v4f*)(xp + 4);
            v4f x2 = *(const v4f*)(xp + 8), x3 = *(const v4f*)(xp + 12);
            v8h h0, h1;
#pragma unroll
            for (int i = 0; i < 4; ++i) {
                h0[i] = (_Float16)x0[i]; h0[4 + i] = (_Float16)x1[i];
                h1[i] = (_Float16)x2[i]; h1[4 + i] = (_Float16)x3[i];
            }
            *(v8h*)&As[ar * LDA + ah]     = h0;
            *(v8h*)&As[ar * LDA + ah + 8] = h1;
        }
        {   // stage W tile 32x64 transposed -> Bt[h][k]
            v8h hb;
#pragma unroll
            for (int i = 0; i < 8; ++i)
                hb[i] = (_Float16)W[((size_t)nh * 1024 + kbase + bd + i) * 64 + bh];
            *(v8h*)&Bt[bh * LDA + bd] = hb;
        }
        __syncthreads();

        v16h a0 = load_a_frag(As + (wm * 32)      * LDA, LDA, lane);
        v16h a1 = load_a_frag(As + (wm * 32 + 16) * LDA, LDA, lane);
        v16h b0 = load_b_frag(Bt + (wn * 32)      * LDA, LDA, lane);
        v16h b1 = load_b_frag(Bt + (wn * 32 + 16) * LDA, LDA, lane);
        acc[0][0] = wmma_f16(a0, b0, acc[0][0]);
        acc[0][1] = wmma_f16(a0, b1, acc[0][1]);
        acc[1][0] = wmma_f16(a1, b0, acc[1][0]);
        acc[1][1] = wmma_f16(a1, b1, acc[1][1]);
    }

    const int nl = lane & 15, hi8 = (lane & 16) >> 1;   // C layout: row r + hi8, col nl
#pragma unroll
    for (int mi = 0; mi < 2; ++mi)
#pragma unroll
        for (int ni = 0; ni < 2; ++ni) {
            const int hcol = wn * 32 + ni * 16 + nl;
            const float bv = bias[nh * 64 + hcol];
#pragma unroll
            for (int r = 0; r < 8; ++r) {
                const int row = rowbase + wm * 32 + mi * 16 + r + hi8;
                const int b = row >> 11, s = row & 2047;
                const _Float16 v = (_Float16)(acc[mi][ni][r] + bv);
                if (w == 2)
                    Vt[((size_t)(b * 16 + nh) * 64 + hcol) * 2048 + s] = v;       // [b,n,h,s]
                else {
                    _Float16* dst = (w == 0) ? Qws : Kws;
                    dst[((size_t)(b * 16 + nh) * 2048 + s) * 64 + hcol] = v;      // [b,n,s,h]
                }
            }
        }
}

// ---------------------------------------------------------------------------
// Kernel 2: causal flash attention. One wave = 16 query rows of one (b, head).
// ---------------------------------------------------------------------------
__global__ __launch_bounds__(256) void flash_attn_kernel(
    const _Float16* __restrict__ Qws, const _Float16* __restrict__ Kws,
    const _Float16* __restrict__ Vt, _Float16* __restrict__ Zws)
{
    __shared__ __align__(16) _Float16 Pl[8 * 16 * LDA];   // per-wave P scratch

    const int tid = threadIdx.x, lane = tid & 31, wave = tid >> 5;
    const int bx = blockIdx.x;
    const int qt = bx & 15, nh = (bx >> 4) & 15, b = bx >> 8;
    const size_t hoff = (size_t)(b * 16 + nh) * 2048 * 64;
    const _Float16* Qh = Qws + hoff;
    const _Float16* Kh = Kws + hoff;
    const _Float16* Vh = Vt  + hoff;
    _Float16* Pw = &Pl[wave * 16 * LDA];

    const int qbase = qt * 128 + wave * 16;
    const int nl = lane & 15;
    const int hi8  = (lane & 16) >> 1;     // 0 or 8
    const int bsel = lane & 16;            // 0 or 16 (B-frag k-offset)

    // Q fragments for this 16-row tile, pre-scaled by 1/sqrt(64)=0.125 (exact in f16)
    v16h aq0 = load_a_frag(Qh + (size_t)qbase * 64, 64, lane);
    v16h aq1 = load_a_frag(Qh + (size_t)qbase * 64 + 32, 64, lane);
#pragma unroll
    for (int i = 0; i < 16; ++i) {
        aq0[i] = aq0[i] * (_Float16)0.125f;
        aq1[i] = aq1[i] * (_Float16)0.125f;
    }

    float mrow[8], lrow[8];
    v8f acc[4] = {};
#pragma unroll
    for (int r = 0; r < 8; ++r) { mrow[r] = -1e30f; lrow[r] = 0.0f; }

    const int nkt = (qbase + 16 + 31) >> 5;          // 32-key tiles (causal bound)
    for (int kt = 0; kt < nkt; ++kt) {
        const int kb = kt * 32;
        // --- scores: two 16-column subtiles, contraction over H=64 (2 wmma each)
        v16h k00 = ld16h(Kh + (size_t)(kb + nl) * 64 + bsel);
        v16h k01 = ld16h(Kh + (size_t)(kb + nl) * 64 + 32 + bsel);
        v16h k10 = ld16h(Kh + (size_t)(kb + 16 + nl) * 64 + bsel);
        v16h k11 = ld16h(Kh + (size_t)(kb + 16 + nl) * 64 + 32 + bsel);
        v8f zf = {};
        v8f s0 = wmma_f16(aq0, k00, zf); s0 = wmma_f16(aq1, k01, s0);
        v8f s1 = wmma_f16(aq0, k10, zf); s1 = wmma_f16(aq1, k11, s1);

        const int col0 = kb + nl, col1 = col0 + 16;
        __builtin_amdgcn_wave_barrier();   // keep prior P-frag load before new stores
#pragma unroll
        for (int r = 0; r < 8; ++r) {
            const int row = qbase + r + hi8;
            float x0 = (col0 <= row) ? s0[r] : -1e30f;
            float x1 = (col1 <= row) ? s1[r] : -1e30f;
            float mx = fmaxf(x0, x1);
            mx = fmaxf(mx, __shfl_xor(mx, 1, 32));
            mx = fmaxf(mx, __shfl_xor(mx, 2, 32));
            mx = fmaxf(mx, __shfl_xor(mx, 4, 32));
            mx = fmaxf(mx, __shfl_xor(mx, 8, 32));
            const float mnew = fmaxf(mrow[r], mx);
            const float sc = __expf(mrow[r] - mnew);
            const float p0 = __expf(x0 - mnew);
            const float p1 = __expf(x1 - mnew);
            float su = p0 + p1;
            su += __shfl_xor(su, 1, 32);
            su += __shfl_xor(su, 2, 32);
            su += __shfl_xor(su, 4, 32);
            su += __shfl_xor(su, 8, 32);
            lrow[r] = lrow[r] * sc + su;
            mrow[r] = mnew;
#pragma unroll
            for (int t = 0; t < 4; ++t) acc[t][r] *= sc;
            // transpose P (C layout) -> LDS row-major [16][32]
            Pw[(r + hi8) * LDA + nl]      = (_Float16)p0;
            Pw[(r + hi8) * LDA + nl + 16] = (_Float16)p1;
        }
        __builtin_amdgcn_wave_barrier();   // stores before A-frag reload (intra-wave LDS in-order)
        v16h ap = load_a_frag(Pw, LDA, lane);
#pragma unroll
        for (int ht = 0; ht < 4; ++ht) {   // O += P(16x32) x V(32x16) per h-subtile
            v16h bv = ld16h(Vh + (size_t)(ht * 16 + nl) * 2048 + kb + bsel);
            acc[ht] = wmma_f16(ap, bv, acc[ht]);
        }
    }

    // epilogue: z[b, q, n, h] as f16 [4096, 1024]
#pragma unroll
    for (int ht = 0; ht < 4; ++ht) {
        const float* a = (const float*)&acc[ht];
#pragma unroll
        for (int r = 0; r < 8; ++r) {
            const int qpos = qbase + r + hi8;
            Zws[((size_t)(b * 2048 + qpos)) * 1024 + nh * 64 + ht * 16 + nl] =
                (_Float16)(a[r] / lrow[r]);
        }
    }
}

// ---------------------------------------------------------------------------
// Kernel 3: output projection. Z[4096,1024]f16 x W_O[1024,1024]f32 -> out f32
// A-tile staged via Tensor Data Mover (TDM) when available: byte-exact f16
// copy with pad_interval=64B / pad_amount=16B reproducing the LDA=40 layout.
// ---------------------------------------------------------------------------
__global__ __launch_bounds__(256) void out_proj_kernel(
    const _Float16* __restrict__ Z, const float* __restrict__ Wo,
    const float* __restrict__ bO, float* __restrict__ out)
{
    __shared__ __align__(16) _Float16 As[128 * LDA];
    __shared__ __align__(16) _Float16 Bt[64 * LDA];

    const int tid = threadIdx.x, lane = tid & 31, wave = tid >> 5;
    const int colbase = blockIdx.x * 64;
    const int rowbase = blockIdx.y * 128;
    const int wm = wave & 3, wn = wave >> 2;

    v8f acc[2][2] = {};
    const int ar = tid >> 1, ah = (tid & 1) * 16;
    const int bd = tid & 63, bk = (tid >> 6) * 8;

#ifdef USE_TDM
    const unsigned lds_base =
        (unsigned)(unsigned long long)(__attribute__((address_space(3))) _Float16*)&As[0];
    const unsigned long long gbase =
        (unsigned long long)(size_t)(Z + (size_t)rowbase * 1024);
#endif

    for (int kk = 0; kk < 32; ++kk) {
        const int kbase = kk * 32;
        __syncthreads();
#ifdef USE_TDM
        if (wave == 0) {   // TDM: DMA Z tile 128x32 f16 -> As with 16B row pad
            const unsigned long long ga = gbase + (unsigned long long)kbase * 2;
            v4u g0;
            g0[0] = 1u;                                        // count=1, user mode
            g0[1] = lds_base;                                  // lds_addr
            g0[2] = (unsigned)(ga & 0xffffffffu);              // global_addr[31:0]
            g0[3] = (unsigned)((ga >> 32) & 0x01ffffffu) | 0x80000000u; // addr[56:32], type=2
            v8i g1;
            g1[0] = (int)((1u << 16)   // data_size = 2 bytes
                        | (1u << 20)   // pad_enable
                        | (3u << 22)   // pad_interval: after every 16 DWORDs (64B)
                        | (3u << 25)); // pad_amount: 4 DWORDs (16B)
            g1[1] = (int)(1024u << 16);    // tensor_dim0 = 1024 elems (row length)
            g1[2] = (int)(128u << 16);     // tensor_dim1 = 128 rows
            g1[3] = (int)(32u << 16);      // tile_dim0 = 32 elems (64B)
            g1[4] = 128;                   // tile_dim1 = 128, tile_dim2 = 0
            g1[5] = 1024;                  // tensor_dim0_stride = 1024 elems
            g1[6] = 0;
            g1[7] = 0;
            v4i g2 = {0, 0, 0, 0}, g3 = {0, 0, 0, 0};
            v8i g4 = {0, 0, 0, 0, 0, 0, 0, 0};   // extra descriptor words (clang-23 6-arg form)
            __builtin_amdgcn_tensor_load_to_lds(g0, g1, g2, g3, g4, 0);
            __builtin_amdgcn_s_wait_tensorcnt(0);
        }
#else
        {   // stage Z tile (already f16)
            const _Float16* zp = Z + (size_t)(rowbase + ar) * 1024 + kbase + ah;
            *(v8h*)&As[ar * LDA + ah]     = *(const v8h*)zp;
            *(v8h*)&As[ar * LDA + ah + 8] = *(const v8h*)(zp + 8);
        }
#endif
        {   // stage W_O tile transposed: Bt[d][c]  (f32 -> f16)
            v8h hb;
#pragma unroll
            for (int i = 0; i < 8; ++i)
                hb[i] = (_Float16)Wo[(size_t)(kbase + bk + i) * 1024 + colbase + bd];
            *(v8h*)&Bt[bd * LDA + bk] = hb;
        }
        __syncthreads();

        v16h a0 = load_a_frag(As + (wm * 32)      * LDA, LDA, lane);
        v16h a1 = load_a_frag(As + (wm * 32 + 16) * LDA, LDA, lane);
        v16h b0 = load_b_frag(Bt + (wn * 32)      * LDA, LDA, lane);
        v16h b1 = load_b_frag(Bt + (wn * 32 + 16) * LDA, LDA, lane);
        acc[0][0] = wmma_f16(a0, b0, acc[0][0]);
        acc[0][1] = wmma_f16(a0, b1, acc[0][1]);
        acc[1][0] = wmma_f16(a1, b0, acc[1][0]);
        acc[1][1] = wmma_f16(a1, b1, acc[1][1]);
    }

    const int nl = lane & 15, hi8 = (lane & 16) >> 1;
#pragma unroll
    for (int mi = 0; mi < 2; ++mi)
#pragma unroll
        for (int ni = 0; ni < 2; ++ni) {
            const int col = colbase + wn * 32 + ni * 16 + nl;
            const float bv = bO[col];
#pragma unroll
            for (int r = 0; r < 8; ++r) {
                const int row = rowbase + wm * 32 + mi * 16 + r + hi8;
                out[(size_t)row * 1024 + col] = acc[mi][ni][r] + bv;
            }
        }
}

extern "C" void kernel_launch(void* const* d_in, const int* in_sizes, int n_in,
                              void* d_out, int out_size, void* d_ws, size_t ws_size,
                              hipStream_t stream) {
    const float* x  = (const float*)d_in[0];
    const float* WQ = (const float*)d_in[1];
    const float* WK = (const float*)d_in[2];
    const float* WV = (const float*)d_in[3];
    const float* WO = (const float*)d_in[4];
    const float* bQ = (const float*)d_in[5];
    const float* bK = (const float*)d_in[6];
    const float* bV = (const float*)d_in[7];
    const float* bO = (const float*)d_in[8];
    float* out = (float*)d_out;

    const size_t SLAB = (size_t)4096 * 1024;       // halves per tensor (8 MB)
    _Float16* Qws = (_Float16*)d_ws;               // [b,n,s,h]
    _Float16* Kws = Qws + SLAB;                    // [b,n,s,h]
    _Float16* Vt  = Kws + SLAB;                    // [b,n,h,s]
    _Float16* Zws = Vt  + SLAB;                    // [b*s, n*h]

    qkv_proj_kernel<<<dim3(48, 32), 256, 0, stream>>>(x, WQ, WK, WV, bQ, bK, bV, Qws, Kws, Vt);
    flash_attn_kernel<<<dim3(512), 256, 0, stream>>>(Qws, Kws, Vt, Zws);
    out_proj_kernel<<<dim3(16, 32), 256, 0, stream>>>(Zws, WO, bO, out);
}
